// TestModule_31877247271320
// MI455X (gfx1250) — matrix operand order repaired and checked by reading, not compiled, not run on hardware
//
#include <hip/hip_runtime.h>

typedef __bf16 bf16;
typedef __attribute__((ext_vector_type(16))) __bf16 v16bf;
typedef __attribute__((ext_vector_type(8)))  __bf16 bf16x8;
typedef __attribute__((ext_vector_type(4)))  __bf16 bf16x4;
typedef __attribute__((ext_vector_type(8)))  float  v8f;
typedef __attribute__((ext_vector_type(4)))  float  f32x4;

static constexpr int HIDDEN = 2048;
static constexpr int FFN    = 8192;
static constexpr int KB     = 32;    // K per step (one bf16 WMMA depth)
static constexpr int BM     = 128;   // block tile M
static constexpr int BN     = 256;   // block tile N
static constexpr int LDK    = 40;    // padded K stride in LDS (bf16) = 80B; 16B-aligned chunks

// Native bf16 converts (v_cvt_bf16_f32 / v_cvt_f32_bf16 on gfx1250, RNE).
__device__ __forceinline__ bf16  f2bf(float x) { return (bf16)x; }
__device__ __forceinline__ float bf2f(bf16 b)  { return (float)b; }

__global__ __launch_bounds__(256)
void grouped_gemm_bf16x3_kernel(const float* __restrict__ hs,
                                const float* __restrict__ wts,
                                float* __restrict__ out) {
    __shared__ __align__(32) bf16 Ah[BM * LDK];
    __shared__ __align__(32) bf16 Al[BM * LDK];
    __shared__ __align__(32) bf16 Bh[BN * LDK];
    __shared__ __align__(32) bf16 Bl[BN * LDK];

    const int tid      = threadIdx.x;
    const int row_base = blockIdx.y * BM;   // 0..16383, also output row base
    const int col_base = blockIdx.x * BN;   // 0..8191

    // Static ragged-group lookup (offsets are multiples of 128; no straddling).
    constexpr int OFF[9] = {0, 1024, 2560, 4608, 6656, 9216, 11264, 14336, 16384};
    int g = 0;
#pragma unroll
    for (int i = 1; i < 8; ++i)
        if (row_base >= OFF[i]) g = i;
    const float* __restrict__ w = wts + (size_t)g * HIDDEN * FFN;

    // 8 waves -> 2 (M) x 4 (N); wave tile = 64M x 64N.
    const int lane = tid & 31;
    const int wave = tid >> 5;
    const int wm   = wave & 1;   // M position (64 rows each)
    const int wn   = wave >> 1;  // N position (64 cols each)
    const int half = lane >> 4;  // 0/1
    const int lrow = lane & 15;

    // Global->LDS loader coordinates.
    const int a_col  = (tid & 7) * 4;   // 0..28 (float4 along K)
    const int a_row0 = tid >> 3;        // 0..31 (x4 passes -> 128 rows)
    const int b_n    = tid;             // one column per thread (256 cols)

    v8f acc[16] = {};   // [mt*4 + nt], each a 16x16 f32 tile

    for (int kb = 0; kb < HIDDEN; kb += KB) {
        // ---- Stage A: 128 rows x 32 K, hi/lo bf16 into LDS ----
#pragma unroll
        for (int p = 0; p < 4; ++p) {
            const int row = a_row0 + 32 * p;
            f32x4 v = *reinterpret_cast<const f32x4*>(
                hs + (size_t)(row_base + row) * HIDDEN + kb + a_col);
            bf16x4 h, l;
#pragma unroll
            for (int i = 0; i < 4; ++i) {
                float x = v[i];
                bf16 hb = f2bf(x);
                h[i]    = hb;
                l[i]    = f2bf(x - bf2f(hb));
            }
            *reinterpret_cast<bf16x4*>(&Ah[row * LDK + a_col]) = h;
            *reinterpret_cast<bf16x4*>(&Al[row * LDK + a_col]) = l;
        }
        // ---- Stage B: 32 K x 256 cols, transposed to [n][k], hi/lo ----
#pragma unroll
        for (int q = 0; q < 8; ++q) {
            bf16x4 h, l;
#pragma unroll
            for (int i = 0; i < 4; ++i) {
                const int k = 4 * q + i;
                float x = w[(size_t)(kb + k) * FFN + col_base + b_n];
                bf16 hb = f2bf(x);
                h[i]    = hb;
                l[i]    = f2bf(x - bf2f(hb));
            }
            *reinterpret_cast<bf16x4*>(&Bh[b_n * LDK + 4 * q]) = h;
            *reinterpret_cast<bf16x4*>(&Bl[b_n * LDK + 4 * q]) = l;
        }
        __syncthreads();

        // ---- Fragments per ISA VGPR layouts ----
        // A 16-bit 16x32: lane<16 -> K {0..7, 16..23}; lane>=16 -> K {8..15, 24..31}
        v16bf afh[4], afl[4];
#pragma unroll
        for (int mt = 0; mt < 4; ++mt) {
            const int row = wm * 64 + mt * 16 + lrow;
            {
                const bf16* b0 = &Ah[row * LDK];
                bf16x8 c0 = *reinterpret_cast<const bf16x8*>(b0 + half * 8);
                bf16x8 c1 = *reinterpret_cast<const bf16x8*>(b0 + 16 + half * 8);
                afh[mt] = __builtin_shufflevector(c0, c1,
                    0,1,2,3,4,5,6,7,8,9,10,11,12,13,14,15);
            }
            {
                const bf16* b0 = &Al[row * LDK];
                bf16x8 c0 = *reinterpret_cast<const bf16x8*>(b0 + half * 8);
                bf16x8 c1 = *reinterpret_cast<const bf16x8*>(b0 + 16 + half * 8);
                afl[mt] = __builtin_shufflevector(c0, c1,
                    0,1,2,3,4,5,6,7,8,9,10,11,12,13,14,15);
            }
        }
        // B 16-bit 32x16: lane holds N=lane&15, K {half*8..half*8+7, 16+half*8..16+half*8+7} (matches A)
        v16bf bfh[4], bfl[4];
#pragma unroll
        for (int nt = 0; nt < 4; ++nt) {
            const int col = wn * 64 + nt * 16 + lrow;
            {
                const bf16* b0 = &Bh[col * LDK];
                bf16x8 c0 = *reinterpret_cast<const bf16x8*>(b0 + half * 8);
                bf16x8 c1 = *reinterpret_cast<const bf16x8*>(b0 + 16 + half * 8);
                bfh[nt] = __builtin_shufflevector(c0, c1,
                    0,1,2,3,4,5,6,7,8,9,10,11,12,13,14,15);
            }
            {
                const bf16* b0 = &Bl[col * LDK];
                bf16x8 c0 = *reinterpret_cast<const bf16x8*>(b0 + half * 8);
                bf16x8 c1 = *reinterpret_cast<const bf16x8*>(b0 + 16 + half * 8);
                bfl[nt] = __builtin_shufflevector(c0, c1,
                    0,1,2,3,4,5,6,7,8,9,10,11,12,13,14,15);
            }
        }

        // ---- bf16x3 accumulation: AhBh + AhBl + AlBh (48 WMMA / wave / step) ----
#pragma unroll
        for (int mt = 0; mt < 4; ++mt) {
#pragma unroll
            for (int nt = 0; nt < 4; ++nt) {
                const int i = mt * 4 + nt;
                acc[i] = __builtin_amdgcn_wmma_f32_16x16x32_bf16(
                    false, afh[mt], false, bfh[nt], (short)0, acc[i], false, false);
                acc[i] = __builtin_amdgcn_wmma_f32_16x16x32_bf16(
                    false, afh[mt], false, bfl[nt], (short)0, acc[i], false, false);
                acc[i] = __builtin_amdgcn_wmma_f32_16x16x32_bf16(
                    false, afl[mt], false, bfh[nt], (short)0, acc[i], false, false);
            }
        }
        __syncthreads();
    }

    // ---- Store: C/D 16x16 f32 layout: VGPR r, lane -> M = r + 8*half, N = lrow ----
#pragma unroll
    for (int mt = 0; mt < 4; ++mt) {
#pragma unroll
        for (int nt = 0; nt < 4; ++nt) {
            const int i    = mt * 4 + nt;
            const int row0 = row_base + wm * 64 + mt * 16 + half * 8;
            const int col0 = col_base + wn * 64 + nt * 16 + lrow;
#pragma unroll
            for (int r = 0; r < 8; ++r) {
                out[(size_t)(row0 + r) * FFN + col0] = acc[i][r];
            }
        }
    }
}

extern "C" void kernel_launch(void* const* d_in, const int* in_sizes, int n_in,
                              void* d_out, int out_size, void* d_ws, size_t ws_size,
                              hipStream_t stream) {
    (void)in_sizes; (void)n_in; (void)out_size; (void)d_ws; (void)ws_size;
    const float* hs  = (const float*)d_in[0];   // [16384, 2048] f32
    const float* wts = (const float*)d_in[1];   // [8, 2048, 8192] f32
    float* out       = (float*)d_out;           // [16384, 8192] f32

    dim3 grid(FFN / BN, 16384 / BM);            // 32 x 128 blocks
    dim3 block(256);
    grouped_gemm_bf16x3_kernel<<<grid, block, 0, stream>>>(hs, wts, out);
}
